// InverseMaskModel_29283087024598
// MI455X (gfx1250) — compile-verified, matrix operand order checked by reading, not run on hardware
//
#include <hip/hip_runtime.h>
#include <hip/hip_bf16.h>

typedef __attribute__((ext_vector_type(16))) _Float16 v16h;
typedef __attribute__((ext_vector_type(8)))  _Float16 v8h;
typedef __attribute__((ext_vector_type(8)))  float    v8f;

#define G_GRID 7
#define SG     12
#define PATCH  49
#define PF     144       // patch features
#define A_DIM  18
#define BATCH  8192
#define IMG    7056      // 84*84
#define ROWS   (BATCH * PATCH)   // 401408
#define WAVES_PER_BLK 4
#define SMEM_PER_WAVE (32 * 320 + 32 * 256)   // 18432 halves = 36864 B

// ---------------------------------------------------------------------------
// WMMA fragment geometry (V_WMMA_F32_16X16X32_F16, wave32):
//   A: 16x32 (MxK), lane<16 -> M=lane, halves h: k = (h<8? h : h+8)
//                   lane>=16 -> M=lane-16, halves h: k = (h<8? h+8 : h+16)
//   B: 32x16 (KxN), same striping with N = lane&15.
//   C/D f32 16x16: lane&15 = N, vgpr r -> M = r + (lane<16 ? 0 : 8)
// Packed weights stored as [kt][nt][lane(32)][half(16)] so each lane's B
// fragment is 32 contiguous bytes.
// ---------------------------------------------------------------------------
__device__ inline int kmap(int l, int h) {
  return (l < 16) ? ((h < 8) ? h : h + 8)
                  : ((h < 8) ? h + 8 : h + 16);
}

__global__ void pack_weights(const float* __restrict__ W, _Float16* __restrict__ P,
                             int Kreal, int Nreal, int NT, int KT) {
  int idx = blockIdx.x * blockDim.x + threadIdx.x;
  int total = KT * NT * 512;
  if (idx >= total) return;
  int h  = idx & 15;
  int l  = (idx >> 4) & 31;
  int nt = (idx >> 9) % NT;
  int kt = idx / (NT * 512);
  int n = nt * 16 + (l & 15);
  int k = kt * 32 + kmap(l, h);
  float v = (k < Kreal && n < Nreal) ? W[k * Nreal + n] : 0.0f;
  P[idx] = (_Float16)v;
}

__device__ inline v16h load_afrag(const _Float16* p) {
  v8h x0 = *(const v8h*)(p);        // halves 0..7  : k = koff + 0..7
  v8h x1 = *(const v8h*)(p + 16);   // halves 8..15 : k = koff + 16..23
  return __builtin_shufflevector(x0, x1,
         0,1,2,3,4,5,6,7,8,9,10,11,12,13,14,15);
}

// Two 16x16 output tiles (rows 0..15 and 16..31) sharing each B fragment.
template <int KT, int NT>
__device__ inline void tile_mm2(const _Float16* aBase, int lda,
                                const _Float16* __restrict__ bPack,
                                int nt, int lane, v8f& acc0, v8f& acc1) {
  int m    = lane & 15;
  int koff = (lane < 16) ? 0 : 8;
  const _Float16* ar0 = aBase + m * lda + koff;
  const _Float16* ar1 = aBase + (m + 16) * lda + koff;
#pragma unroll
  for (int kt = 0; kt < KT; ++kt) {
    v16h bv = *(const v16h*)(bPack + ((size_t)(kt * NT + nt) * 32 + lane) * 16);
    v16h a0 = load_afrag(ar0 + kt * 32);
    acc0 = __builtin_amdgcn_wmma_f32_16x16x32_f16(
               false, a0, false, bv, (short)0, acc0, false, false);
    v16h a1 = load_afrag(ar1 + kt * 32);
    acc1 = __builtin_amdgcn_wmma_f32_16x16x32_f16(
               false, a1, false, bv, (short)0, acc1, false, false);
  }
}

// Full layer for 32 rows: out = relu(A * W + b), f16 result to LDS
template <int KT, int NT>
__device__ inline void layer_to_lds(const _Float16* aBase, int lda,
                                    const _Float16* __restrict__ bPack,
                                    const float* __restrict__ bias, int Nreal,
                                    _Float16* outBuf, int ldo, int lane) {
  for (int nt = 0; nt < NT; ++nt) {
    v8f acc0 = {}, acc1 = {};
    tile_mm2<KT, NT>(aBase, lda, bPack, nt, lane, acc0, acc1);
    int n = nt * 16 + (lane & 15);
    if (n < Nreal) {
      float bv = bias[n];
      int mb = (lane < 16) ? 0 : 8;
#pragma unroll
      for (int r = 0; r < 8; ++r) {
        float v0 = acc0[r] + bv;
        v0 = v0 > 0.0f ? v0 : 0.0f;
        outBuf[(mb + r) * ldo + n] = (_Float16)v0;
        float v1 = acc1[r] + bv;
        v1 = v1 > 0.0f ? v1 : 0.0f;
        outBuf[(16 + mb + r) * ldo + n] = (_Float16)v1;
      }
    }
  }
}

// ---------------------------------------------------------------------------
// Kernel 2: fused patch-gather + both MLPs.
// 4 waves/block, 32 rows/wave. Waves advance through the same weight-fragment
// stream in lock-step (layer barriers), so B fragments are reused via WGP$.
// ---------------------------------------------------------------------------
__global__ void __launch_bounds__(32 * WAVES_PER_BLK)
mlp_kernel(const float* __restrict__ last, const float* __restrict__ now,
           const _Float16* __restrict__ We1p, const _Float16* __restrict__ We2p,
           const _Float16* __restrict__ We3p, const _Float16* __restrict__ Wa1p,
           const _Float16* __restrict__ Wa2p, const _Float16* __restrict__ Wa3p,
           const float* __restrict__ be1, const float* __restrict__ be2,
           const float* __restrict__ be3, const float* __restrict__ ba1,
           const float* __restrict__ ba2, const float* __restrict__ ba3,
           float* __restrict__ e_ws, float* __restrict__ alpha_ws) {
  // Per wave: bufA 32x320 (e_in + K-pad, later layer-2 out), bufH 32x256
  // (hidden1); alpha hiddens alias into bufH (dead before e-L1 writes it).
  __shared__ __align__(32) _Float16 smem[WAVES_PER_BLK * SMEM_PER_WAVE];
  int lane = threadIdx.x & 31;
  int wave = threadIdx.x >> 5;
  _Float16* bufA  = smem + wave * SMEM_PER_WAVE;  // 10240 halves
  _Float16* bufH  = bufA + 10240;                 // 8192 halves
  _Float16* bufS1 = bufH;                         // 32 x 64
  _Float16* bufS2 = bufH + 2048;                  // 32 x 64
  int row0 = (blockIdx.x * WAVES_PER_BLK + wave) * 32;

  // ---- gather patch pixels, build e_in = [now-last, now] as f16 ----
  for (int idx = lane; idx < 32 * PF; idx += 32) {
    int m = idx / PF, k = idx % PF;
    int row = row0 + m;
    int b = row / PATCH, p = row % PATCH;
    int pi = p / G_GRID, pj = p % G_GRID;
    int r = k / SG, c = k % SG;
    size_t off = (size_t)b * IMG + (size_t)(pi * SG + r) * 84 + (pj * SG + c);
    float nw = now[off]  * (1.0f / 255.0f);
    float ls = last[off] * (1.0f / 255.0f);
    bufA[m * 320 + k]      = (_Float16)(nw - ls);
    bufA[m * 320 + PF + k] = (_Float16)nw;
  }
  for (int idx = lane; idx < 32 * 32; idx += 32) {
    int m = idx / 32, k = idx % 32;
    bufA[m * 320 + 288 + k] = (_Float16)0.0f;   // K-pad for alpha layer 1
  }
  __syncthreads();

  // ---- alpha MLP: 144 -> 64 -> 64 -> 1  (input = now half of e_in) ----
  layer_to_lds<5, 4>(bufA + PF, 320, Wa1p, ba1, 64, bufS1, 64, lane);
  __syncthreads();
  layer_to_lds<2, 4>(bufS1, 64, Wa2p, ba2, 64, bufS2, 64, lane);
  __syncthreads();
  {
    v8f acc0 = {}, acc1 = {};
    tile_mm2<2, 1>(bufS2, 64, Wa3p, 0, lane, acc0, acc1);
    if ((lane & 15) == 0) {               // column n == 0 of padded N=16
      float bv = ba3[0];
      int mb = (lane < 16) ? 0 : 8;
#pragma unroll
      for (int r = 0; r < 8; ++r) {
        alpha_ws[row0 + mb + r]      = acc0[r] + bv;
        alpha_ws[row0 + 16 + mb + r] = acc1[r] + bv;
      }
    }
  }
  __syncthreads();   // alpha hiddens (aliased in bufH) dead after this point

  // ---- e MLP: 288 -> 256 -> 128 -> 18 ----
  layer_to_lds<9, 16>(bufA, 320, We1p, be1, 256, bufH, 256, lane);
  __syncthreads();
  layer_to_lds<8, 8>(bufH, 256, We2p, be2, 128, bufA, 320, lane);  // reuse bufA
  __syncthreads();
  for (int nt = 0; nt < 2; ++nt) {        // N padded 18 -> 32
    v8f acc0 = {}, acc1 = {};
    tile_mm2<4, 2>(bufA, 320, We3p, nt, lane, acc0, acc1);
    int n = nt * 16 + (lane & 15);
    if (n < A_DIM) {
      float bv = be3[n];
      int mb = (lane < 16) ? 0 : 8;
#pragma unroll
      for (int r = 0; r < 8; ++r) {
        e_ws[(size_t)(row0 + mb + r) * A_DIM + n]      = acc0[r] + bv;
        e_ws[(size_t)(row0 + 16 + mb + r) * A_DIM + n] = acc1[r] + bv;
      }
    }
  }
}

__global__ void zero_loss(float* p) { *p = 0.0f; }

// ---------------------------------------------------------------------------
// Kernel 3: per-batch epilogue, one wave per batch element.
// ---------------------------------------------------------------------------
__global__ void __launch_bounds__(256)
combine_kernel(const float* __restrict__ e_ws, const float* __restrict__ alpha_ws,
               float* __restrict__ out1, float* __restrict__ loss,
               float* __restrict__ out_each) {
  int lane = threadIdx.x & 31;
  int wave = threadIdx.x >> 5;
  int b = blockIdx.x * 8 + wave;
  __shared__ float sm[8 * 192];
  float* alpha = sm + wave * 192;       // [64]
  float* plog  = alpha + 64;            // [64] per-patch logsumexp
  const float* eb = e_ws + (size_t)b * PATCH * A_DIM;
  const float* ab = alpha_ws + (size_t)b * PATCH;

  // softmax over 49 alpha logits (lanes hold p = lane and p = lane+32)
  float x0 = ab[lane];
  float x1 = (lane + 32 < PATCH) ? ab[lane + 32] : -1e30f;
  float mx = fmaxf(x0, x1);
#pragma unroll
  for (int m = 16; m; m >>= 1) mx = fmaxf(mx, __shfl_xor(mx, m, 32));
  float e0 = __expf(x0 - mx);
  float e1 = (lane + 32 < PATCH) ? __expf(x1 - mx) : 0.0f;
  float s = e0 + e1;
#pragma unroll
  for (int m = 16; m; m >>= 1) s += __shfl_xor(s, m, 32);
  float inv = 1.0f / s;
  float a0 = e0 * inv, a1 = e1 * inv;
  alpha[lane] = a0;
  if (lane + 32 < PATCH) alpha[lane + 32] = a1;

  float ent = a0 * __logf(a0);
  if (lane + 32 < PATCH) ent += a1 * __logf(a1);
#pragma unroll
  for (int m = 16; m; m >>= 1) ent += __shfl_xor(ent, m, 32);
  if (lane == 0) atomicAdd(loss, ent * (1.0f / (float)BATCH));
  __syncthreads();

  // weighted logits + log_softmax over A=18
  float sa = 0.0f;
  if (lane < A_DIM)
    for (int p = 0; p < PATCH; ++p) sa += alpha[p] * eb[p * A_DIM + lane];
  float v = (lane < A_DIM) ? sa : -1e30f;
  float vmax = v;
#pragma unroll
  for (int m = 16; m; m >>= 1) vmax = fmaxf(vmax, __shfl_xor(vmax, m, 32));
  float se = (lane < A_DIM) ? __expf(sa - vmax) : 0.0f;
#pragma unroll
  for (int m = 16; m; m >>= 1) se += __shfl_xor(se, m, 32);
  if (lane < A_DIM) out1[(size_t)b * A_DIM + lane] = sa - vmax - __logf(se);

  // per-patch log_softmax: first logsumexp per patch
  for (int p = lane; p < PATCH; p += 32) {
    float m2 = -1e30f;
    for (int a = 0; a < A_DIM; ++a) m2 = fmaxf(m2, eb[p * A_DIM + a]);
    float s2 = 0.0f;
    for (int a = 0; a < A_DIM; ++a) s2 += __expf(eb[p * A_DIM + a] - m2);
    plog[p] = m2 + __logf(s2);
  }
  __syncthreads();
  for (int idx = lane; idx < PATCH * A_DIM; idx += 32) {
    int p = idx / A_DIM, a = idx % A_DIM;
    out_each[((size_t)b * PATCH + p) * A_DIM + a] = eb[p * A_DIM + a] - plog[p];
  }
}

// ---------------------------------------------------------------------------
extern "C" void kernel_launch(void* const* d_in, const int* in_sizes, int n_in,
                              void* d_out, int out_size, void* d_ws, size_t ws_size,
                              hipStream_t stream) {
  const float* last = (const float*)d_in[0];
  const float* now  = (const float*)d_in[1];
  const float* We1 = (const float*)d_in[2];
  const float* be1 = (const float*)d_in[3];
  const float* We2 = (const float*)d_in[4];
  const float* be2 = (const float*)d_in[5];
  const float* We3 = (const float*)d_in[6];
  const float* be3 = (const float*)d_in[7];
  const float* Wa1 = (const float*)d_in[8];
  const float* ba1 = (const float*)d_in[9];
  const float* Wa2 = (const float*)d_in[10];
  const float* ba2 = (const float*)d_in[11];
  const float* Wa3 = (const float*)d_in[12];
  const float* ba3 = (const float*)d_in[13];

  char* ws = (char*)d_ws;
  _Float16* We1p = (_Float16*)(ws + 0);        // 9*16*512 halves = 147456 B
  _Float16* We2p = (_Float16*)(ws + 147456);   // 8*8*512        = 65536 B
  _Float16* We3p = (_Float16*)(ws + 212992);   // 4*2*512        = 8192 B
  _Float16* Wa1p = (_Float16*)(ws + 221184);   // 5*4*512        = 20480 B
  _Float16* Wa2p = (_Float16*)(ws + 241664);   // 2*4*512        = 8192 B
  _Float16* Wa3p = (_Float16*)(ws + 249856);   // 2*1*512        = 2048 B
  float* e_ws     = (float*)(ws + 251904);     // ROWS*18 f32
  float* alpha_ws = (float*)(ws + 251904 + (size_t)ROWS * A_DIM * 4);

  float* out1 = (float*)d_out;                 // [B,18]
  float* loss = out1 + (size_t)BATCH * A_DIM;  // scalar
  float* out_each = loss + 1;                  // [B,49,18]

  auto cdiv = [](int a, int b) { return (a + b - 1) / b; };
  pack_weights<<<cdiv(9*16*512,256),256,0,stream>>>(We1, We1p, 288,256,16,9);
  pack_weights<<<cdiv(8*8*512,256),256,0,stream>>>(We2, We2p, 256,128, 8,8);
  pack_weights<<<cdiv(4*2*512,256),256,0,stream>>>(We3, We3p, 128, 18, 2,4);
  pack_weights<<<cdiv(5*4*512,256),256,0,stream>>>(Wa1, Wa1p, 144, 64, 4,5);
  pack_weights<<<cdiv(2*4*512,256),256,0,stream>>>(Wa2, Wa2p,  64, 64, 4,2);
  pack_weights<<<cdiv(2*1*512,256),256,0,stream>>>(Wa3, Wa3p,  64,  1, 1,2);

  mlp_kernel<<<ROWS / (32 * WAVES_PER_BLK), 32 * WAVES_PER_BLK, 0, stream>>>(
      last, now, We1p, We2p, We3p, Wa1p, Wa2p, Wa3p,
      be1, be2, be3, ba1, ba2, ba3, e_ws, alpha_ws);

  zero_loss<<<1, 1, 0, stream>>>(loss);
  combine_kernel<<<BATCH / 8, 256, 0, stream>>>(e_ws, alpha_ws, out1, loss, out_each);
}
